// Model_66142496358596
// MI455X (gfx1250) — matrix operand order repaired and checked by reading, not compiled, not run on hardware
//
#include <hip/hip_runtime.h>
#include <hip/hip_bf16.h>

// ---------------------------------------------------------------------------
// Seq2seq LSTM (B=32, T_IN=128, T_OUT=64, H=2048, IO=1) for MI455X / gfx1250.
//
// Strategy:
//  * bf16 weights + v_wmma_f32_16x16x32_bf16 (fp32 accumulate).
//  * 128 persistent workgroups; each owns 16 h-columns -> its 4x16 gate rows
//    of W_hh live in LDS (256 KB, B-fragment layout) for the whole phase.
//  * Per step: 8 waves (2 M-tiles x 4 gates) each run 64 WMMAs over K=2048.
//  * h exchanged via double-buffered bf16 global buffer; device-wide
//    sense-reversal barrier (agent-scope atomics) once per time step.
//  * c-state is block-private in LDS. Decoder y_t = partial dots atomicAdd'ed
//    into d_out (pre-seeded with fc_b by init kernel) in the same phase as h_t.
// ---------------------------------------------------------------------------

#define NB      32      // batch
#define TIN     128
#define TOUT    64
#define HD      2048
#define NBLK    128     // persistent workgroups (one per WGP)
#define COLS    16      // h-columns owned per workgroup
#define THREADS 256     // 8 waves of 32
#define KBLK    (HD / 32)   // 64 k-blocks of 32 for the bf16 WMMA

typedef __attribute__((ext_vector_type(16))) __bf16 v16bf;
typedef __attribute__((ext_vector_type(8)))  float  v8f;

// LDS carve-up (bytes)
#define LDS_W_BYTES   (4 * KBLK * 32 * 16 * 2)      // 262144: W slice, B-frag order
#define LDS_GATE_OFF  (LDS_W_BYTES)                  // float[4][32][16] = 8192
#define LDS_C_OFF     (LDS_GATE_OFF + 4*NB*COLS*4)   // float[32][16]    = 2048
#define LDS_WIH_OFF   (LDS_C_OFF + NB*COLS*4)        // float[4][16]
#define LDS_B_OFF     (LDS_WIH_OFF + 64*4)           // float[4][16]
#define LDS_FCW_OFF   (LDS_B_OFF + 64*4)             // float[16]
#define SMEM_BYTES    (LDS_FCW_OFF + 16*4)           // ~272.9 KB < 320 KB

// workspace layout (bytes)
#define WS_CNT 0
#define WS_GEN 4
#define WS_H0  256
#define WS_H1  (256 + NB * HD * 2)

__device__ __forceinline__ float sigm(float v) {
    return 1.0f / (1.0f + __expf(-v));
}

__device__ __forceinline__ void grid_barrier(unsigned* cnt, unsigned* gen) {
    __syncthreads();
    if (threadIdx.x == 0) {
        __threadfence();  // release our h / out stores to device scope
        unsigned g0 = __hip_atomic_load(gen, __ATOMIC_RELAXED, __HIP_MEMORY_SCOPE_AGENT);
        unsigned arrived =
            __hip_atomic_fetch_add(cnt, 1u, __ATOMIC_ACQ_REL, __HIP_MEMORY_SCOPE_AGENT);
        if (arrived == NBLK - 1) {
            __hip_atomic_store(cnt, 0u, __ATOMIC_RELAXED, __HIP_MEMORY_SCOPE_AGENT);
            __hip_atomic_fetch_add(gen, 1u, __ATOMIC_RELEASE, __HIP_MEMORY_SCOPE_AGENT);
        } else {
            while (__hip_atomic_load(gen, __ATOMIC_ACQUIRE, __HIP_MEMORY_SCOPE_AGENT) == g0) {
                __builtin_amdgcn_s_sleep(2);
            }
        }
    }
    __syncthreads();
    __threadfence();  // acquire: invalidate stale lines before reading peers' h
}

// Load this block's W_hh slice into LDS in B-fragment order:
//   lw[((g*KBLK + kb)*16 + nn)*32 + kk] = W_hh[g*HD + j0 + nn][kb*32 + kk]
// so a lane's B-frag (n = lane&15, k = 16*half .. +15) is one contiguous 32-byte DS read.
__device__ void load_weights(__bf16* lw, float* wih, float* bias,
                             const float* __restrict__ W_hh,
                             const float* __restrict__ W_ih,
                             const float* __restrict__ bvec,
                             int j0, int tid) {
    for (int r = 0; r < 4 * COLS; ++r) {        // 64 gate rows owned
        int g  = r >> 4;
        int nn = r & 15;
        const float* row = W_hh + (size_t)(g * HD + j0 + nn) * HD;  // coalesced reads
        for (int k = tid; k < HD; k += THREADS) {
            lw[(((size_t)g * KBLK + (k >> 5)) * 16 + nn) * 32 + (k & 31)] = (__bf16)row[k];
        }
    }
    if (tid < 64) {
        int g = tid >> 4, nn = tid & 15;
        wih[tid]  = W_ih[g * HD + j0 + nn];     // IO == 1
        bias[tid] = bvec[g * HD + j0 + nn];
    }
}

extern "C" __global__ void __launch_bounds__(THREADS, 1)
seq2seq_lstm_kernel(const float* __restrict__ x,
                    const float* __restrict__ enc_W_ih,
                    const float* __restrict__ enc_W_hh,
                    const float* __restrict__ enc_b,
                    const float* __restrict__ dec_W_ih,
                    const float* __restrict__ dec_W_hh,
                    const float* __restrict__ dec_b,
                    const float* __restrict__ fc_W,
                    float* __restrict__ out,
                    __bf16* __restrict__ h0buf,
                    __bf16* __restrict__ h1buf,
                    unsigned* cnt, unsigned* gen) {
    extern __shared__ char smem[];
    __bf16* lw      = (__bf16*)smem;
    float*  gateBuf = (float*)(smem + LDS_GATE_OFF);   // [4][32][16]
    float*  cState  = (float*)(smem + LDS_C_OFF);      // [32][16]
    float*  wih     = (float*)(smem + LDS_WIH_OFF);    // [4][16]
    float*  bias    = (float*)(smem + LDS_B_OFF);      // [4][16]
    float*  fcw     = (float*)(smem + LDS_FCW_OFF);    // [16]

    const int tid = threadIdx.x;
    const int j0  = blockIdx.x * COLS;

    for (int i = tid; i < NB * COLS; i += THREADS) cState[i] = 0.0f;
    if (tid < COLS) fcw[tid] = fc_W[j0 + tid];
    load_weights(lw, wih, bias, enc_W_hh, enc_W_ih, enc_b, j0, tid);
    __syncthreads();

    const int wave  = tid >> 5;
    const int lane  = tid & 31;
    const int mtile = wave >> 2;    // 0..1 : batch rows 0-15 / 16-31
    const int g     = wave & 3;     // gate index i,f,g,o
    const int mrow  = lane & 15;
    const int half  = lane >> 4;

    __bf16* hprev = h0buf;
    __bf16* hnext = h1buf;

    for (int step = 0; step < TIN + TOUT; ++step) {
        const bool dec = (step >= TIN);
        const int  t   = dec ? (step - TIN) : step;

        if (dec && t == 0) {   // phase switch: swap in decoder weights
            __syncthreads();
            load_weights(lw, wih, bias, dec_W_hh, dec_W_ih, dec_b, j0, tid);
            __syncthreads();
        }

        // ---- gates_slice = h_prev @ W_slice^T via bf16 WMMA (fp32 acc) ----
        v8f acc = {0.f, 0.f, 0.f, 0.f, 0.f, 0.f, 0.f, 0.f};
        const uint4*  arow  = (const uint4*)(hprev + (size_t)(mtile * 16 + mrow) * HD);
        const __bf16* bbase = lw + ((size_t)g * KBLK * 16 + (lane & 15)) * 32 + half * 16;
        #pragma unroll 4
        for (int kb = 0; kb < KBLK; ++kb) {
            // A frag (contiguous k order, same as B): lanes 0-15: K 0-15,
            // lanes 16-31: K 16-31 -> two contiguous 16B loads per lane.
            union { uint4 u[2]; v16bf v; } a;
            a.u[0] = arow[kb * 4 + 2 * half];
            a.u[1] = arow[kb * 4 + 2 * half + 1];
            // B frag: lane&15 = N column, 16 bf16 K-values (16*half..) contiguous in LDS.
            v16bf b = *(const v16bf*)(bbase + (size_t)kb * 32 * 16);
            acc = __builtin_amdgcn_wmma_f32_16x16x32_bf16(
                false, a.v, false, b, (short)0, acc, false, false);
        }
        // C/D layout: VGPR v -> M = mtile*16 + half*8 + v, N = lane&15.
        {
            float* gb = gateBuf + ((g * NB + mtile * 16 + half * 8) * COLS) + (lane & 15);
            #pragma unroll
            for (int v = 0; v < 8; ++v) gb[v * COLS] = acc[v];
        }
        __syncthreads();

        // ---- pointwise LSTM cell on the 32x16 owned slice ----
        #pragma unroll
        for (int s = 0; s < 2; ++s) {
            int p  = tid + s * THREADS;       // 512 (b,nn) pairs
            int b  = p >> 4;
            int nn = p & 15;
            float inp;
            if (!dec)            inp = x[b * TIN + t];
            else if (t == 0)     inp = x[b * TIN + (TIN - 1)];
            else                 inp = out[b * TOUT + (t - 1)];  // y_{t-1}
            float gi = gateBuf[(0 * NB + b) * COLS + nn] + bias[0 * COLS + nn] + inp * wih[0 * COLS + nn];
            float gf = gateBuf[(1 * NB + b) * COLS + nn] + bias[1 * COLS + nn] + inp * wih[1 * COLS + nn];
            float gg = gateBuf[(2 * NB + b) * COLS + nn] + bias[2 * COLS + nn] + inp * wih[2 * COLS + nn];
            float go = gateBuf[(3 * NB + b) * COLS + nn] + bias[3 * COLS + nn] + inp * wih[3 * COLS + nn];
            float c  = sigm(gf) * cState[b * COLS + nn] + sigm(gi) * tanhf(gg);
            float h  = sigm(go) * tanhf(c);
            cState[b * COLS + nn] = c;
            hnext[(size_t)b * HD + j0 + nn] = (__bf16)h;
            if (dec) atomicAdd(&out[b * TOUT + t], h * fcw[nn]);  // partial fc dot
        }

        grid_barrier(cnt, gen);
        __bf16* tmp = hprev; hprev = hnext; hnext = tmp;
    }
}

// Deterministic per-call init: barrier counters, h buffers, out = fc_b.
extern "C" __global__ void seq2seq_init(float* __restrict__ out,
                                        const float* __restrict__ fc_b,
                                        unsigned* __restrict__ ws_u32) {
    int i      = blockIdx.x * blockDim.x + threadIdx.x;
    int stride = gridDim.x * blockDim.x;
    if (i == 0) { ws_u32[WS_CNT / 4] = 0u; ws_u32[WS_GEN / 4] = 0u; }
    unsigned* h = ws_u32 + WS_H0 / 4;
    for (int k = i; k < (2 * NB * HD * 2) / 4; k += stride) h[k] = 0u;  // both h bufs
    float fb = fc_b[0];
    for (int k = i; k < NB * TOUT; k += stride) out[k] = fb;
}

extern "C" void kernel_launch(void* const* d_in, const int* in_sizes, int n_in,
                              void* d_out, int out_size, void* d_ws, size_t ws_size,
                              hipStream_t stream) {
    (void)in_sizes; (void)n_in; (void)out_size; (void)ws_size;
    const float* x        = (const float*)d_in[0];
    // d_in[1] = target (unused)
    const float* enc_W_ih = (const float*)d_in[2];
    const float* enc_W_hh = (const float*)d_in[3];
    const float* enc_b    = (const float*)d_in[4];
    const float* dec_W_ih = (const float*)d_in[5];
    const float* dec_W_hh = (const float*)d_in[6];
    const float* dec_b    = (const float*)d_in[7];
    const float* fc_W     = (const float*)d_in[8];
    const float* fc_b     = (const float*)d_in[9];
    // d_in[10] = teacher_forcing_ratio == 0 (unused)

    float* out = (float*)d_out;
    char*  ws  = (char*)d_ws;
    unsigned* cnt = (unsigned*)(ws + WS_CNT);
    unsigned* gen = (unsigned*)(ws + WS_GEN);
    __bf16*   h0  = (__bf16*)(ws + WS_H0);
    __bf16*   h1  = (__bf16*)(ws + WS_H1);

    // Opt in to >64 KB dynamic LDS (gfx1250 WGP has 320 KB).
    (void)hipFuncSetAttribute((const void*)seq2seq_lstm_kernel,
                              hipFuncAttributeMaxDynamicSharedMemorySize, SMEM_BYTES);

    seq2seq_init<<<64, 256, 0, stream>>>(out, fc_b, (unsigned*)d_ws);
    seq2seq_lstm_kernel<<<NBLK, THREADS, SMEM_BYTES, stream>>>(
        x, enc_W_ih, enc_W_hh, enc_b, dec_W_ih, dec_W_hh, dec_b, fc_W,
        out, h0, h1, cnt, gen);
}
